// SeparableConv2DTransposeONNX_40407052320819
// MI455X (gfx1250) — matrix operand order repaired and checked by reading, not compiled, not run on hardware
//
#include <hip/hip_runtime.h>

typedef __attribute__((ext_vector_type(16))) __bf16 v16bf;
typedef __attribute__((ext_vector_type(8)))  __bf16 v8bf;
typedef __attribute__((ext_vector_type(4)))  __bf16 v4bf;
typedef __attribute__((ext_vector_type(8)))  float  v8f;

#define BPITCH 264               // bf16 elems per B row (256 + 8 pad)
#define APITCH 264               // bf16 elems per A row
#define LDS_ELEMS (256*BPITCH + 32*APITCH)
#define TILES_PER_BLOCK 4

extern "C" __global__ void __launch_bounds__(256)
sepconv_tconv_wmma(const float* __restrict__ in,    // [16,64,64,256]
                   const float* __restrict__ dwk,   // [4,4,256,1]
                   const float* __restrict__ pwk,   // [256,256] (K x N)
                   const float* __restrict__ bias,  // [256]
                   float* __restrict__ out,         // [16,128,128,256]
                   int nTiles)
{
  extern __shared__ __bf16 lds[];
  __bf16* ldsB = lds;                 // [256][BPITCH]  B transposed: [n][k], k contiguous
  __bf16* ldsA = lds + 256*BPITCH;    // [32][APITCH]   A tile: [m][k], k contiguous

  const int tid = threadIdx.x;

  // ---------------- Stage pointwise weights (f32 -> bf16, transposed) into LDS, once ----------
  {
    int sub = tid;
    #pragma unroll
    for (int it = 0; it < 16; ++it, sub += 256) {     // 4096 4x4 sub-blocks total
      const int k0 = (sub >> 6) << 2;                 // 0..252
      const int n0 = (sub & 63) << 2;                 // 0..252
      const float4 r0 = *reinterpret_cast<const float4*>(pwk + (k0 + 0)*256 + n0);
      const float4 r1 = *reinterpret_cast<const float4*>(pwk + (k0 + 1)*256 + n0);
      const float4 r2 = *reinterpret_cast<const float4*>(pwk + (k0 + 2)*256 + n0);
      const float4 r3 = *reinterpret_cast<const float4*>(pwk + (k0 + 3)*256 + n0);
      v4bf q0, q1, q2, q3;  // transpose 4x4, native bf16 converts (v_cvt_pk_bf16_f32)
      q0[0]=(__bf16)r0.x; q0[1]=(__bf16)r1.x; q0[2]=(__bf16)r2.x; q0[3]=(__bf16)r3.x;
      q1[0]=(__bf16)r0.y; q1[1]=(__bf16)r1.y; q1[2]=(__bf16)r2.y; q1[3]=(__bf16)r3.y;
      q2[0]=(__bf16)r0.z; q2[1]=(__bf16)r1.z; q2[2]=(__bf16)r2.z; q2[3]=(__bf16)r3.z;
      q3[0]=(__bf16)r0.w; q3[1]=(__bf16)r1.w; q3[2]=(__bf16)r2.w; q3[3]=(__bf16)r3.w;
      *reinterpret_cast<v4bf*>(&ldsB[(n0+0)*BPITCH + k0]) = q0;
      *reinterpret_cast<v4bf*>(&ldsB[(n0+1)*BPITCH + k0]) = q1;
      *reinterpret_cast<v4bf*>(&ldsB[(n0+2)*BPITCH + k0]) = q2;
      *reinterpret_cast<v4bf*>(&ldsB[(n0+3)*BPITCH + k0]) = q3;
    }
  }

  // wave / lane decomposition (wave32)
  const int lane = tid & 31;
  const int wav  = tid >> 5;        // 0..7
  const int wm   = wav & 1;         // M half  (16 rows)
  const int wn   = wav >> 1;        // N quarter (64 cols)
  const int lrow = lane & 15;
  const int hlf  = lane >> 4;

  // depthwise-stage decomposition
  const int m  = tid >> 3;          // pixel within tile 0..31
  const int c0 = (tid & 7) << 5;    // channel block 0..224

  for (int t = 0; t < TILES_PER_BLOCK; ++t) {
    const int tile = blockIdx.x * TILES_PER_BLOCK + t;
    if (tile >= nTiles) break;                  // uniform per block
    const int x0 = (tile & 3) << 5;             // 4 tiles of 32 px per image row
    const int y  = (tile >> 2) & 127;
    const int b  = tile >> 9;

    __syncthreads();   // previous compute (and initial B staging) done before A rewrite / B read

    // -------- Stage depthwise transposed-conv tile (32 px x 256 ch) -> bf16 LDS_A ------------
    {
      const int x = x0 + m;
      int iy[2], ky[2], ix[2], kx[2];
      if (y & 1) { iy[0]=(y-1)>>1; ky[0]=2; iy[1]=(y+1)>>1; ky[1]=0; }
      else       { iy[0]=(y>>1)-1; ky[0]=3; iy[1]=y>>1;     ky[1]=1; }
      if (x & 1) { ix[0]=(x-1)>>1; kx[0]=2; ix[1]=(x+1)>>1; kx[1]=0; }
      else       { ix[0]=(x>>1)-1; kx[0]=3; ix[1]=x>>1;     kx[1]=1; }

      const float* tin[4]; const float* twt[4]; int ntap = 0;
      #pragma unroll
      for (int r = 0; r < 2; ++r)
        #pragma unroll
        for (int s = 0; s < 2; ++s)
          if (iy[r] >= 0 && iy[r] < 64 && ix[s] >= 0 && ix[s] < 64) {
            tin[ntap] = in  + (((b*64 + iy[r])*64) + ix[s]) * 256;
            twt[ntap] = dwk + (ky[r]*4 + kx[s]) * 256;
            ++ntap;
          }

      for (int c = c0; c < c0 + 32; c += 4) {
        float a0 = 0.f, a1 = 0.f, a2 = 0.f, a3 = 0.f;
        for (int tp = 0; tp < ntap; ++tp) {
          const float4 xi = *reinterpret_cast<const float4*>(tin[tp] + c);
          const float4 wi = *reinterpret_cast<const float4*>(twt[tp] + c);
          a0 += xi.x * wi.x; a1 += xi.y * wi.y; a2 += xi.z * wi.z; a3 += xi.w * wi.w;
        }
        v4bf q; q[0]=(__bf16)a0; q[1]=(__bf16)a1; q[2]=(__bf16)a2; q[3]=(__bf16)a3;
        *reinterpret_cast<v4bf*>(&ldsA[m*APITCH + c]) = q;
      }
    }

    __syncthreads();

    // -------- Prefetch a slab of the next tile's input (global_prefetch_b8 hint) -------------
    {
      const int tile2 = tile + 1;
      if (tile2 < nTiles) {
        const int x0n = (tile2 & 3) << 5;
        const int yn  = (tile2 >> 2) & 127;
        const int bn  = tile2 >> 9;
        const int iyn = (yn >> 1) < 63 ? (yn >> 1) : 63;
        const float* p = in + (((bn*64 + iyn)*64) + x0n + lane) * 256 + wav * 32;
        __builtin_prefetch(p, 0, 1);
      }
    }

    // -------- GEMM: (32 x 256) x (256 x 256) via v_wmma_f32_16x16x32_bf16 --------------------
    v8f acc[4] = {};
    const __bf16* aRow = ldsA + (wm*16 + lrow) * APITCH;

    #pragma unroll
    for (int k0 = 0; k0 < 256; k0 += 32) {
      // A fragment 16x32: lanes 0-15 hold K {k0+0..7, k0+16..23}, lanes 16-31 hold {+8..15,+24..31}
      v8bf alo = *reinterpret_cast<const v8bf*>(aRow + k0 + hlf*8);
      v8bf ahi = *reinterpret_cast<const v8bf*>(aRow + k0 + 16 + hlf*8);
      v16bf a = __builtin_shufflevector(alo, ahi, 0,1,2,3,4,5,6,7,8,9,10,11,12,13,14,15);
      #pragma unroll
      for (int f = 0; f < 4; ++f) {
        // B fragment 32x16: N = lane%16, lanes 0-15 K {k0+0..7, k0+16..23}, lanes 16-31 K {+8..15,+24..31}
        const __bf16* bRow = ldsB + ((wn*4 + f)*16 + lrow) * BPITCH + k0 + hlf*8;
        v8bf blo = *reinterpret_cast<const v8bf*>(bRow);
        v8bf bhi = *reinterpret_cast<const v8bf*>(bRow + 16);
        v16bf bm = __builtin_shufflevector(blo, bhi, 0,1,2,3,4,5,6,7,8,9,10,11,12,13,14,15);
        acc[f] = __builtin_amdgcn_wmma_f32_16x16x32_bf16(
            /*neg_a=*/false, a, /*neg_b=*/false, bm,
            /*c_mod=*/(short)0, acc[f], /*reuse_a=*/false, /*reuse_b=*/false);
      }
    }

    // -------- Epilogue: bias + relu, streaming (non-temporal) stores -------------------------
    #pragma unroll
    for (int f = 0; f < 4; ++f) {
      const int col = (wn*4 + f)*16 + lrow;
      const float bv = bias[col];
      float* op = out + (((b*128 + y)*128) + x0 + wm*16 + hlf*8) * 256 + col;
      #pragma unroll
      for (int r = 0; r < 8; ++r) {
        float v = acc[f][r] + bv;
        v = v > 0.f ? v : 0.f;
        __builtin_nontemporal_store(v, op + r*256);
      }
    }
  }
}

extern "C" void kernel_launch(void* const* d_in, const int* in_sizes, int n_in,
                              void* d_out, int out_size, void* d_ws, size_t ws_size,
                              hipStream_t stream) {
  const float* in   = (const float*)d_in[0];
  const float* dwk  = (const float*)d_in[1];
  const float* pwk  = (const float*)d_in[2];
  const float* bias = (const float*)d_in[3];
  float* out = (float*)d_out;
  (void)d_ws; (void)ws_size; (void)n_in; (void)out_size;

  const int Bsz    = in_sizes[0] / (64 * 64 * 256);    // batch (16)
  const int nTiles = Bsz * 128 * 4;                    // 32-pixel tiles
  const int grid   = (nTiles + TILES_PER_BLOCK - 1) / TILES_PER_BLOCK;

  const size_t shmem = (size_t)LDS_ELEMS * sizeof(__bf16);  // ~149 KB, fits 2/WGP (320 KB)
  (void)hipFuncSetAttribute((const void*)sepconv_tconv_wmma,
                            hipFuncAttributeMaxDynamicSharedMemorySize, (int)shmem);

  hipLaunchKernelGGL(sepconv_tconv_wmma, dim3(grid), dim3(256), shmem, stream,
                     in, dwk, pwk, bias, out, nTiles);
}
